// TenerAttention_59167469469718
// MI455X (gfx1250) — matrix operand order repaired and checked by reading, not compiled, not run on hardware
//
#include <hip/hip_runtime.h>
#include <hip/hip_bf16.h>

// ---------------------------------------------------------------------------
// TENER attention for MI455X (gfx1250), wave32 + WMMA bf16 + async-LDS copy.
//
// scores[b,h,i,j] = (q+u)·k + (q+v)·Rrel[j-i]   (Toeplitz R, XL-style fold)
// Flash-softmax streaming over 64x64 tiles; all contractions on
// v_wmma_f32_16x16x32_bf16 (f32 accumulate).  Inputs converted to bf16 once;
// GEMM B tiles double-buffered in LDS via global_load_async_to_lds_b128 with
// a manually unrolled (static-parity) software pipeline.
//
// Workspace layout (bf16 scratch, ~57 MB):
//   [0,8M)    xb    (b*s, 1024)
//   [8,10M)   Wqb   [10,12M) Wkb   [12,14M) Wvb   [14,16M) Wob
//   [16,24M)  qu    (b,h,s,64)
//   [24,32M)  qv
//   [32,40M)  k     (b,h,s,64)
//   [40,48M)  vT    (b,h,64,s)   <- V pre-transposed for P@V B-fragments
//   [48,56M)  ctx   (b,s,1024)
//   [56M..)   Rrel  (2176 x 64), rows >= 2047 zero (window padding)
// ---------------------------------------------------------------------------

typedef __attribute__((ext_vector_type(16))) __bf16 bf16x16;
typedef __attribute__((ext_vector_type(8)))  __bf16 bf16x8;
typedef __attribute__((ext_vector_type(8)))  float  f32x8;
typedef __attribute__((ext_vector_type(4)))  float  f32x4;
typedef __attribute__((ext_vector_type(4)))  unsigned int u32x4;

__device__ __forceinline__ f32x8 wmma_bf16(bf16x16 a, bf16x16 b, f32x8 c) {
    // (neg_a, A, neg_b, B, c_mod, C, reuse_a, reuse_b)
    return __builtin_amdgcn_wmma_f32_16x16x32_bf16(false, a, false, b,
                                                   (short)0, c, false, false);
}

// A fragment (16x32 bf16, M x K), row-major source with leading dim `ld`.
// Lane L: row = L%16; K chunks [kb..kb+7] and [kb+16..kb+23], kb = (L/16)*8.
__device__ __forceinline__ bf16x16 load_a_frag(const __bf16* tile, int ld) {
    int lane = threadIdx.x & 31;
    int r    = lane & 15;
    int kb   = (lane >> 4) << 3;
    const __bf16* p = tile + r * ld + kb;
    union { bf16x16 v; u32x4 q[2]; } u;
    u.q[0] = *(const u32x4*)(p);
    u.q[1] = *(const u32x4*)(p + 16);
    return u.v;
}

// B fragment (32x16 bf16, K x N) sourced from row-major (N x K) storage:
// lane L: col n = L%16; K chunks [kb..kb+7] and [kb+16..kb+23], kb = (L/16)*8.
__device__ __forceinline__ bf16x16 load_b_frag(const __bf16* tile, int ld) {
    int lane = threadIdx.x & 31;
    int n    = lane & 15;
    int kb   = (lane >> 4) << 3;
    const __bf16* p = tile + n * ld + kb;
    union { bf16x16 v; u32x4 q[2]; } u;
    u.q[0] = *(const u32x4*)(p);
    u.q[1] = *(const u32x4*)(p + 16);
    return u.v;
}

// Async copy of one 16-byte chunk global -> LDS; split waits for pipelining.
__device__ __forceinline__ void async_copy_b128(unsigned lds_off,
                                                const __bf16* gsrc) {
    asm volatile("global_load_async_to_lds_b128 %0, %1, off"
                 :: "v"(lds_off), "v"(gsrc) : "memory");
}
__device__ __forceinline__ void async_wait0() {
    asm volatile("s_wait_asynccnt 0" ::: "memory");
}
__device__ __forceinline__ void async_wait1() {
    asm volatile("s_wait_asynccnt 1" ::: "memory");
}

// One pipelined GEMM K-step: optionally prefetch tile i+1 (async B copy +
// A fragment loads), wait for tile i, then 2x2 WMMAs from static buffer bsc.
template<bool PREFETCH>
__device__ __forceinline__ void gemm_step(
    const __bf16* bsc, unsigned lds_next, const __bf16* wsrc_next,
    const __bf16* ag_next, bf16x16& a0n, bf16x16& a1n,
    bf16x16 a0, bf16x16 a1, int wc,
    f32x8& acc00, f32x8& acc01, f32x8& acc10, f32x8& acc11) {
    if (PREFETCH) {
        async_copy_b128(lds_next, wsrc_next);
        a0n = load_a_frag(ag_next, 1024);
        a1n = load_a_frag(ag_next + (size_t)16 * 1024, 1024);
        async_wait1();                 // tile i landed (one copy in flight)
    } else {
        async_wait0();
    }
    __syncthreads();                   // all chunks of tile i landed
    bf16x16 b0 = load_b_frag(bsc + (wc * 32) * 32, 32);
    bf16x16 b1 = load_b_frag(bsc + (wc * 32 + 16) * 32, 32);
    acc00 = wmma_bf16(a0, b0, acc00);
    acc01 = wmma_bf16(a0, b1, acc01);
    acc10 = wmma_bf16(a1, b0, acc10);
    acc11 = wmma_bf16(a1, b1, acc11);
    __syncthreads();                   // reads done before buffer overwrite
}

// ---------------------------------------------------------------------------
// Kernel 0: f32 -> bf16 conversion (one streaming pass, 8 elems/thread).
// ---------------------------------------------------------------------------
__global__ __launch_bounds__(256) void tener_cvt(const float* __restrict__ s,
                                                 __bf16* __restrict__ d,
                                                 int n8) {
    int i = blockIdx.x * 256 + threadIdx.x;
    if (i >= n8) return;
    f32x4 a = *(const f32x4*)(s + (size_t)i * 8);
    f32x4 b = *(const f32x4*)(s + (size_t)i * 8 + 4);
    bf16x8 o;
    for (int k = 0; k < 4; ++k) { o[k] = (__bf16)a[k]; o[k + 4] = (__bf16)b[k]; }
    *(bf16x8*)(d + (size_t)i * 8) = o;
}

// ---------------------------------------------------------------------------
// Kernel 1: fused QKV projection.  z = 0:Q (writes qu,qv), 1:K, 2:V(transposed)
// One block computes a 128-row x 64-col tile; 64 cols == one head.
// ---------------------------------------------------------------------------
__global__ __launch_bounds__(256) void tener_qkv(
    const __bf16* __restrict__ xb,
    const __bf16* __restrict__ Wqb, const __bf16* __restrict__ Wkb,
    const __bf16* __restrict__ Wvb,
    const float* __restrict__ u, const float* __restrict__ v,
    __bf16* __restrict__ qu, __bf16* __restrict__ qv,
    __bf16* __restrict__ kk, __bf16* __restrict__ vT) {
    __shared__ __bf16 Bs[2][64 * 32];
    int rt = blockIdx.x, head = blockIdx.y, z = blockIdx.z;
    const __bf16* W = (z == 0) ? Wqb : (z == 1) ? Wkb : Wvb;
    int tid = threadIdx.x, w = tid >> 5, lane = tid & 31;
    int wr = w & 3, wc = w >> 2, nlo = lane & 15, half = lane >> 4;

    // Per-thread async staging slot: row = tid/4, 16B chunk = tid%4.
    int srow = tid >> 2, schunk = tid & 3;
    unsigned lds0 = (unsigned)(size_t)(&Bs[0][srow * 32 + schunk * 8]);
    unsigned lds1 = (unsigned)(size_t)(&Bs[1][srow * 32 + schunk * 8]);
    const __bf16* wrow = W + (size_t)(head * 64 + srow) * 1024 + schunk * 8;
    const __bf16* ag   = xb + (size_t)(rt * 128 + wr * 32) * 1024;

    async_copy_b128(lds0, wrow);                 // prefetch tile 0
    bf16x16 a0n = load_a_frag(ag, 1024);
    bf16x16 a1n = load_a_frag(ag + (size_t)16 * 1024, 1024);

    f32x8 acc00 = {}, acc01 = {}, acc10 = {}, acc11 = {};
    for (int i = 0; i < 32; i += 2) {            // static-parity pipeline
        bf16x16 a0 = a0n, a1 = a1n;
        gemm_step<true>(Bs[0], lds1, wrow + i * 32 + 32, ag + i * 32 + 32,
                        a0n, a1n, a0, a1, wc, acc00, acc01, acc10, acc11);
        a0 = a0n; a1 = a1n;
        if (i + 2 < 32)
            gemm_step<true>(Bs[1], lds0, wrow + i * 32 + 64, ag + i * 32 + 64,
                            a0n, a1n, a0, a1, wc, acc00, acc01, acc10, acc11);
        else
            gemm_step<false>(Bs[1], 0, nullptr, nullptr,
                             a0n, a1n, a0, a1, wc, acc00, acc01, acc10, acc11);
    }
    for (int rb = 0; rb < 2; ++rb) {
        f32x8 accA = rb ? acc10 : acc00;
        f32x8 accB = rb ? acc11 : acc01;
        for (int r = 0; r < 8; ++r) {
            int row = rt * 128 + wr * 32 + rb * 16 + half * 8 + r;  // [0,4096)
            int b_idx = row >> 10, s_idx = row & 1023;
            for (int cc = 0; cc < 2; ++cc) {
                int d = wc * 32 + cc * 16 + nlo;
                float val = cc ? accB[r] : accA[r];
                size_t bh = (size_t)(b_idx * 16 + head);
                if (z == 0) {
                    size_t o = (bh * 1024 + s_idx) * 64 + d;
                    qu[o] = (__bf16)(val + u[head * 64 + d]);
                    qv[o] = (__bf16)(val + v[head * 64 + d]);
                } else if (z == 1) {
                    kk[(bh * 1024 + s_idx) * 64 + d] = (__bf16)val;
                } else {                              // V transposed (d, s)
                    vT[(bh * 64 + d) * 1024 + s_idx] = (__bf16)val;
                }
            }
        }
    }
}

// ---------------------------------------------------------------------------
// Kernel 2: Rrel[r,d], r in [0,2047): rel = r-1023; rows >= 2047 are zero pad.
// ---------------------------------------------------------------------------
__global__ __launch_bounds__(256) void tener_rrel(__bf16* __restrict__ R) {
    int id = blockIdx.x * 256 + threadIdx.x;
    if (id >= 2176 * 64) return;
    int r = id >> 6, d = id & 63;
    float val = 0.0f;
    if (r < 2047) {
        float rel  = (float)(r - 1023);
        float expo = -(float)(2 * (d >> 1)) * (1.0f / 64.0f);
        float div  = __powf(10000.0f, expo);
        float ang  = rel * div;
        val = (d & 1) ? __cosf(ang) : __sinf(ang);
    }
    R[id] = (__bf16)val;
}

// ---------------------------------------------------------------------------
// Kernel 3: flash attention per (b*h, 64-row i-tile).
// ---------------------------------------------------------------------------
__global__ __launch_bounds__(256) void tener_attn(
    const __bf16* __restrict__ QU, const __bf16* __restrict__ QV,
    const __bf16* __restrict__ K,  const __bf16* __restrict__ VT,
    const __bf16* __restrict__ R,  __bf16* __restrict__ CTX) {
    const int S = 1024, HD = 64;
    int it = blockIdx.x, bh = blockIdx.y;
    int b_idx = bh >> 4, h = bh & 15;
    int i0 = it * 64;
    int tid = threadIdx.x, w = tid >> 5, lane = tid & 31;
    int wr = w & 3, wc = w >> 2, nlo = lane & 15, half = lane >> 4;

    __shared__ __bf16 Pl[64 * 72];        // P tile bf16 (i, j), padded
    __shared__ float  Tl[8][16 * 64];     // per-wave skew-GEMM scratch
    __shared__ float  m_s[64], l_s[64], scale_s[64];
    __shared__ float  pmax[2][64], psum[2][64];

    const __bf16* qu = QU + (size_t)bh * S * HD;
    const __bf16* qv = QV + (size_t)bh * S * HD;
    const __bf16* kp = K  + (size_t)bh * S * HD;
    const __bf16* vt = VT + (size_t)bh * HD * S;   // (d, s) row-major

    if (tid < 64) { m_s[tid] = -1e30f; l_s[tid] = 0.0f; }

    // Persistent A fragments for this wave's 16 query rows.
    bf16x16 aqu0 = load_a_frag(qu + (size_t)(i0 + wr * 16) * HD + 0,  HD);
    bf16x16 aqu1 = load_a_frag(qu + (size_t)(i0 + wr * 16) * HD + 32, HD);
    bf16x16 aqv0 = load_a_frag(qv + (size_t)(i0 + wr * 16) * HD + 0,  HD);
    bf16x16 aqv1 = load_a_frag(qv + (size_t)(i0 + wr * 16) * HD + 32, HD);

    f32x8 ctx0 = {}, ctx1 = {};

    int cmin = wc * 32 - wr * 16 + 48;    // min needed rel-window column
    int c0   = cmin & ~15;                // 16-aligned window base (>= 0)

    for (int jt = 0; jt < 16; ++jt) {
        int j0 = jt * 64;
        __syncthreads();                  // prior P@V reads of Pl complete
        // ---- S tile: (q+u) k^T ----
        f32x8 s0 = {}, s1 = {};
        {
            bf16x16 bk0 = load_b_frag(kp + (size_t)(j0 + wc * 32) * HD + 0,      HD);
            bf16x16 bk1 = load_b_frag(kp + (size_t)(j0 + wc * 32 + 16) * HD + 0, HD);
            s0 = wmma_bf16(aqu0, bk0, s0);
            s1 = wmma_bf16(aqu0, bk1, s1);
            bk0 = load_b_frag(kp + (size_t)(j0 + wc * 32) * HD + 32,      HD);
            bk1 = load_b_frag(kp + (size_t)(j0 + wc * 32 + 16) * HD + 32, HD);
            s0 = wmma_bf16(aqu1, bk0, s0);
            s1 = wmma_bf16(aqu1, bk1, s1);
        }
        // ---- relative-position term: T = qv @ Rrel_slice^T, then skew-gather
        int rbase = j0 - i0 + S - 64;     // Rrel row for c = 0
        for (int ct = 0; ct < 4; ++ct) {
            f32x8 t = {};
            const __bf16* rp = R + (size_t)(rbase + c0 + ct * 16) * HD;
            bf16x16 br0 = load_b_frag(rp + 0,  HD);
            bf16x16 br1 = load_b_frag(rp + 32, HD);
            t = wmma_bf16(aqv0, br0, t);
            t = wmma_bf16(aqv1, br1, t);
            for (int r = 0; r < 8; ++r)
                Tl[w][(half * 8 + r) * 64 + ct * 16 + nlo] = t[r];
        }
        for (int r = 0; r < 8; ++r) {     // add T[a, b-a+63] into S
            int a  = wr * 16 + half * 8 + r;
            int cr = (wc * 32 + nlo) - a + 63 - c0;   // in [0,45]
            s0[r] += Tl[w][(half * 8 + r) * 64 + cr];
            s1[r] += Tl[w][(half * 8 + r) * 64 + cr + 16];
        }
        // ---- online softmax: row max ----
        float pm[8];
        for (int r = 0; r < 8; ++r) {
            float m = fmaxf(s0[r], s1[r]);
            for (int off = 8; off >= 1; off >>= 1)
                m = fmaxf(m, __shfl_xor(m, off, 32));
            pm[r] = m;
        }
        if (nlo == 0)
            for (int r = 0; r < 8; ++r)
                pmax[wc][wr * 16 + half * 8 + r] = pm[r];
        __syncthreads();
        if (tid < 64) {
            float mo = m_s[tid];
            float mn = fmaxf(mo, fmaxf(pmax[0][tid], pmax[1][tid]));
            m_s[tid] = mn;
            scale_s[tid] = __expf(mo - mn);
        }
        __syncthreads();
        // ---- P = exp(S - m), rescale ctx, row sums, stage P as bf16 ----
        float ps[8];
        for (int r = 0; r < 8; ++r) {
            int row  = wr * 16 + half * 8 + r;
            float mn = m_s[row];
            float p0 = __expf(s0[r] - mn);
            float p1 = __expf(s1[r] - mn);
            float sc = scale_s[row];
            ctx0[r] *= sc;
            ctx1[r] *= sc;
            float su = p0 + p1;
            for (int off = 8; off >= 1; off >>= 1)
                su += __shfl_xor(su, off, 32);
            ps[r] = su;
            Pl[row * 72 + wc * 32 + nlo]      = (__bf16)p0;
            Pl[row * 72 + wc * 32 + 16 + nlo] = (__bf16)p1;
        }
        if (nlo == 0)
            for (int r = 0; r < 8; ++r)
                psum[wc][wr * 16 + half * 8 + r] = ps[r];
        __syncthreads();
        if (tid < 64)
            l_s[tid] = l_s[tid] * scale_s[tid] + psum[0][tid] + psum[1][tid];
        // ---- ctx += P @ V : B fragments straight from pre-transposed V ----
        bf16x16 ap0 = load_a_frag(Pl + (wr * 16) * 72 + 0,  72);
        bf16x16 ap1 = load_a_frag(Pl + (wr * 16) * 72 + 32, 72);
        bf16x16 bv0 = load_b_frag(vt + (size_t)(wc * 32) * S + j0 + 0,       S);
        bf16x16 bv1 = load_b_frag(vt + (size_t)(wc * 32 + 16) * S + j0 + 0,  S);
        ctx0 = wmma_bf16(ap0, bv0, ctx0);
        ctx1 = wmma_bf16(ap0, bv1, ctx1);
        bv0 = load_b_frag(vt + (size_t)(wc * 32) * S + j0 + 32,      S);
        bv1 = load_b_frag(vt + (size_t)(wc * 32 + 16) * S + j0 + 32, S);
        ctx0 = wmma_bf16(ap1, bv0, ctx0);
        ctx1 = wmma_bf16(ap1, bv1, ctx1);
    }
    __syncthreads();
    for (int r = 0; r < 8; ++r) {
        int row  = wr * 16 + half * 8 + r;
        float inv = 1.0f / l_s[row];
        int i_g  = i0 + row;
        int d0   = wc * 32 + nlo;
        size_t base = ((size_t)b_idx * 1024 + i_g) * 1024 + h * 64;
        CTX[base + d0]      = (__bf16)(ctx0[r] * inv);
        CTX[base + d0 + 16] = (__bf16)(ctx1[r] * inv);
    }
}

// ---------------------------------------------------------------------------
// Kernel 4: out = ctx @ Wo^T + bo  (fp32 output).  Same pipelined pattern.
// ---------------------------------------------------------------------------
__global__ __launch_bounds__(256) void tener_proj(
    const __bf16* __restrict__ A, const __bf16* __restrict__ Wob,
    const float* __restrict__ bo, float* __restrict__ out) {
    __shared__ __bf16 Bs[2][64 * 32];
    int rt = blockIdx.x, ct = blockIdx.y;
    int tid = threadIdx.x, w = tid >> 5, lane = tid & 31;
    int wr = w & 3, wc = w >> 2, nlo = lane & 15, half = lane >> 4;

    int srow = tid >> 2, schunk = tid & 3;
    unsigned lds0 = (unsigned)(size_t)(&Bs[0][srow * 32 + schunk * 8]);
    unsigned lds1 = (unsigned)(size_t)(&Bs[1][srow * 32 + schunk * 8]);
    const __bf16* wrow = Wob + (size_t)(ct * 64 + srow) * 1024 + schunk * 8;
    const __bf16* ag   = A + (size_t)(rt * 128 + wr * 32) * 1024;

    async_copy_b128(lds0, wrow);
    bf16x16 a0n = load_a_frag(ag, 1024);
    bf16x16 a1n = load_a_frag(ag + (size_t)16 * 1024, 1024);

    f32x8 acc00 = {}, acc01 = {}, acc10 = {}, acc11 = {};
    for (int i = 0; i < 32; i += 2) {
        bf16x16 a0 = a0n, a1 = a1n;
        gemm_step<true>(Bs[0], lds1, wrow + i * 32 + 32, ag + i * 32 + 32,
                        a0n, a1n, a0, a1, wc, acc00, acc01, acc10, acc11);
        a0 = a0n; a1 = a1n;
        if (i + 2 < 32)
            gemm_step<true>(Bs[1], lds0, wrow + i * 32 + 64, ag + i * 32 + 64,
                            a0n, a1n, a0, a1, wc, acc00, acc01, acc10, acc11);
        else
            gemm_step<false>(Bs[1], 0, nullptr, nullptr,
                             a0n, a1n, a0, a1, wc, acc00, acc01, acc10, acc11);
    }
    for (int rb = 0; rb < 2; ++rb) {
        f32x8 accA = rb ? acc10 : acc00;
        f32x8 accB = rb ? acc11 : acc01;
        for (int r = 0; r < 8; ++r) {
            int row = rt * 128 + wr * 32 + rb * 16 + half * 8 + r;
            int col = ct * 64 + wc * 32 + nlo;
            out[(size_t)row * 1024 + col]      = accA[r] + bo[col];
            out[(size_t)row * 1024 + col + 16] = accB[r] + bo[col + 16];
        }
    }
}

// ---------------------------------------------------------------------------
extern "C" void kernel_launch(void* const* d_in, const int* in_sizes, int n_in,
                              void* d_out, int out_size, void* d_ws, size_t ws_size,
                              hipStream_t stream) {
    const float* x  = (const float*)d_in[0];
    const float* Wq = (const float*)d_in[1];
    const float* Wk = (const float*)d_in[2];
    const float* Wv = (const float*)d_in[3];
    const float* Wo = (const float*)d_in[4];
    const float* bo = (const float*)d_in[5];
    const float* u  = (const float*)d_in[6];
    const float* v  = (const float*)d_in[7];

    char* ws = (char*)d_ws;
    const size_t MB = 1024ull * 1024ull;
    __bf16* xb   = (__bf16*)(ws + 0 * MB);    // 8 MB
    __bf16* Wqb  = (__bf16*)(ws + 8 * MB);    // 2 MB each
    __bf16* Wkb  = (__bf16*)(ws + 10 * MB);
    __bf16* Wvb  = (__bf16*)(ws + 12 * MB);
    __bf16* Wob  = (__bf16*)(ws + 14 * MB);
    __bf16* qu   = (__bf16*)(ws + 16 * MB);   // 8 MB each
    __bf16* qv   = (__bf16*)(ws + 24 * MB);
    __bf16* kk   = (__bf16*)(ws + 32 * MB);
    __bf16* vT   = (__bf16*)(ws + 40 * MB);
    __bf16* ctxb = (__bf16*)(ws + 48 * MB);   // 8 MB
    __bf16* Rrel = (__bf16*)(ws + 56 * MB);   // 2176*64*2 = 272 KB

    tener_cvt<<<dim3(2048), 256, 0, stream>>>(x,  xb,  524288);
    tener_cvt<<<dim3(512),  256, 0, stream>>>(Wq, Wqb, 131072);
    tener_cvt<<<dim3(512),  256, 0, stream>>>(Wk, Wkb, 131072);
    tener_cvt<<<dim3(512),  256, 0, stream>>>(Wv, Wvb, 131072);
    tener_cvt<<<dim3(512),  256, 0, stream>>>(Wo, Wob, 131072);

    tener_qkv<<<dim3(32, 16, 3), 256, 0, stream>>>(xb, Wqb, Wkb, Wvb, u, v,
                                                   qu, qv, kk, vT);
    tener_rrel<<<dim3(544), 256, 0, stream>>>(Rrel);
    tener_attn<<<dim3(16, 64), 256, 0, stream>>>(qu, qv, kk, vT, Rrel, ctxb);
    tener_proj<<<dim3(32, 16), 256, 0, stream>>>(ctxb, Wob, bo, (float*)d_out);
}
